// StructuredStateSpaceDualityBranch_49074296324242
// MI455X (gfx1250) — compile-verified, matrix operand order checked by reading, not run on hardware
//
#include <hip/hip_runtime.h>
#include <hip/hip_bf16.h>

// ---------------------------------------------------------------------------
// Problem constants (from the reference)
// ---------------------------------------------------------------------------
#define D_MODEL   1024
#define D_INNER   2048
#define D_STATE   16
#define D_CONV    4
#define DT_RANK   64
#define H_HEADS   8
#define P_DIM     256           // D_INNER / H
#define BATCH     2
#define SEQ       4096
#define NROWS     (BATCH * SEQ) // 8192
#define TOTAL_OUT (2 * D_INNER + DT_RANK + 2 * H_HEADS * D_STATE) // 4416
#define OFF_Z     0
#define OFF_U     (D_INNER)          // 2048
#define OFF_DT    (2 * D_INNER)      // 4096
#define OFF_B     (2 * D_INNER + DT_RANK)                 // 4160
#define OFF_C     (2 * D_INNER + DT_RANK + H_HEADS * D_STATE) // 4288
#define EPSV      1e-6f
#define DT_MIN    1e-4f
#define DT_MAX    1.0f

typedef __attribute__((ext_vector_type(16))) __bf16 v16bf;
typedef __attribute__((ext_vector_type(8)))  __bf16 v8bf;
typedef __attribute__((ext_vector_type(8)))  float  v8f;

// ---------------------------------------------------------------------------
// f32 -> bf16 conversion
// ---------------------------------------------------------------------------
__global__ void cvt_f32_bf16_kernel(const float* __restrict__ in,
                                    __bf16* __restrict__ out, int n) {
  int i = blockIdx.x * blockDim.x + threadIdx.x;
  if (i < n) out[i] = (__bf16)in[i];
}

// ---------------------------------------------------------------------------
// Register-blocked, double-buffered WMMA GEMM: C[M,N] = A[M,K] * B[N,K]^T
// (A,B bf16 row-major, C f32). One wave -> 64x32 macro-tile = 4x2 grid of
// 16x16 WMMA tiles (8 independent accumulators).
//
// K is a template parameter so ALL per-fragment strides (i*16*K elements,
// max 192KB) and k-step offsets fold into the signed 24-bit instruction
// IOFFSET: the whole inner loop needs only TWO 64-bit address registers
// (A base, B base), each bumped once per iteration.
//
// Double buffering: loads always target the fragment buffer the WMMAs are
// NOT reading -> no WMMA->VALU WAR hazard NOPs, and each WMMA group waits
// only on loads issued a full iteration earlier (partial s_wait_loadcnt).
//
// __launch_bounds__(256, 1): working set is ~175 VGPRs (64 acc + 96 frag +
// addressing). Without this the backend caps near 128 VGPRs for occupancy
// and spills accumulator tiles to scratch inside the hot loop; with deep
// double-buffering, 8 waves/WGP hides latency fine, so trade occupancy for
// a spill-free loop.
//
// Fragment layout per CDNA5 ISA 7.12.2 (16-bit A 16x32):
//   lanes 0-15 : row M = lane,    K = k+0..7  and k+16..23
//   lanes 16-31: row M = lane-16, K = k+8..15 and k+24..31
// B (32x16 KxN) loaded column-major == rows of the [N,K] weight matrix.
// ---------------------------------------------------------------------------
#define TM 4
#define TN 2

struct Frags {
  v8bf alo[TM], ahi[TM];
  v8bf blo[TN], bhi[TN];
};

template <int K>
__device__ __forceinline__ void load_frags(Frags& f,
                                           const __bf16* __restrict__ pa,
                                           const __bf16* __restrict__ pb,
                                           int kc) {
#pragma unroll
  for (int i = 0; i < TM; ++i) {
    f.alo[i] = *(const v8bf*)(pa + i * 16 * K + kc);
    f.ahi[i] = *(const v8bf*)(pa + i * 16 * K + kc + 16);
  }
#pragma unroll
  for (int j = 0; j < TN; ++j) {
    f.blo[j] = *(const v8bf*)(pb + j * 16 * K + kc);
    f.bhi[j] = *(const v8bf*)(pb + j * 16 * K + kc + 16);
  }
}

__device__ __forceinline__ void do_wmma(const Frags& f, v8f acc[TM][TN]) {
  v16bf a[TM], b[TN];
#pragma unroll
  for (int i = 0; i < TM; ++i)
#pragma unroll
    for (int e = 0; e < 8; ++e) { a[i][e] = f.alo[i][e]; a[i][e + 8] = f.ahi[i][e]; }
#pragma unroll
  for (int j = 0; j < TN; ++j)
#pragma unroll
    for (int e = 0; e < 8; ++e) { b[j][e] = f.blo[j][e]; b[j][e + 8] = f.bhi[j][e]; }
#pragma unroll
  for (int i = 0; i < TM; ++i)
#pragma unroll
    for (int j = 0; j < TN; ++j)
      acc[i][j] = __builtin_amdgcn_wmma_f32_16x16x32_bf16(
          /*neg_a=*/false, a[i], /*neg_b=*/false, b[j],
          /*c_mod=*/(short)0, acc[i][j],
          /*reuse_a=*/false, /*reuse_b=*/false);
}

template <int K>
__global__ void __launch_bounds__(256, 1)
wmma_gemm_bf16_kernel(const __bf16* __restrict__ A,
                      const __bf16* __restrict__ Bw,
                      float* __restrict__ C,
                      int M, int N) {
  const int wavesPerBlock = blockDim.x >> 5;
  const int wave = blockIdx.x * wavesPerBlock + (threadIdx.x >> 5);
  const int lane = threadIdx.x & 31;
  const int tilesN = N >> 5;                // 32-col macro tiles
  const int tilesM = M >> 6;                // 64-row macro tiles
  const int tmac = wave / tilesN;
  const int tnac = wave % tilesN;
  if (tmac >= tilesM) return;

  const int half = lane >> 4;               // 0 or 1
  const int l16  = lane & 15;
  const int m0 = tmac * 64;
  const int n0 = tnac * 32;
  const int koff = half * 8;

  // one moving base pointer per matrix; all other offsets are immediates
  const __bf16* pa = A + (size_t)(m0 + l16) * K + koff;
  const __bf16* pb = Bw + (size_t)(n0 + l16) * K + koff;

  v8f acc[TM][TN];
#pragma unroll
  for (int i = 0; i < TM; ++i)
#pragma unroll
    for (int j = 0; j < TN; ++j) acc[i][j] = (v8f){};

  // K/32 is even (K in {1024, 2048}) -> exact 2x unrolled pipeline.
  Frags buf0, buf1;
  load_frags<K>(buf0, pa, pb, 0);

  for (int k = 0; k + 64 < K; k += 64) {
    __builtin_prefetch(pa + 512, 0, 1);     // global_prefetch_b8
    __builtin_prefetch(pb + 512, 0, 1);
    load_frags<K>(buf1, pa, pb, 32);
    do_wmma(buf0, acc);
    load_frags<K>(buf0, pa, pb, 64);
    do_wmma(buf1, acc);
    pa += 64;
    pb += 64;
  }
  // epilogue: exactly one k-step pair left
  load_frags<K>(buf1, pa, pb, 32);
  do_wmma(buf0, acc);
  do_wmma(buf1, acc);

  // C/D layout: VGPR r holds M = r (lanes 0-15) / M = r+8 (lanes 16-31)
#pragma unroll
  for (int i = 0; i < TM; ++i) {
#pragma unroll
    for (int j = 0; j < TN; ++j) {
      const int col = n0 + j * 16 + l16;
#pragma unroll
      for (int r = 0; r < 8; ++r) {
        int row = m0 + i * 16 + r + half * 8;
        C[(size_t)row * (size_t)N + col] = acc[i][j][r];
      }
    }
  }
}

// ---------------------------------------------------------------------------
// Causal depthwise conv over seq (width 4) on the u-slice of p.
// u_conv[b,l,c] = conv_b[c] + sum_j conv_w[c,j] * p_u[b, l-3+j, c]
// ---------------------------------------------------------------------------
__global__ void dwconv_kernel(const float* __restrict__ p,
                              const float* __restrict__ conv_w,
                              const float* __restrict__ conv_b,
                              float* __restrict__ u_conv) {
  long idx = (long)blockIdx.x * blockDim.x + threadIdx.x;
  if (idx >= (long)NROWS * D_INNER) return;
  int c  = (int)(idx % D_INNER);
  int bl = (int)(idx / D_INNER);
  int l  = bl & (SEQ - 1);
  float acc = conv_b[c];
#pragma unroll
  for (int j = 0; j < D_CONV; ++j) {
    int lj = l - (D_CONV - 1) + j;
    if (lj >= 0) {
      acc += conv_w[c * D_CONV + j] *
             p[(size_t)(bl - (D_CONV - 1) + j) * TOTAL_OUT + OFF_U + c];
    }
  }
  u_conv[(size_t)bl * D_INNER + c] = acc;
}

// ---------------------------------------------------------------------------
// dt = clip(softplus(dt_h @ W_dt.T + dt_bias), DT_MIN, DT_MAX)
// one thread per (row, head); K = 64
// ---------------------------------------------------------------------------
__global__ void dt_kernel(const float* __restrict__ p,
                          const float* __restrict__ W_dt,
                          const float* __restrict__ dt_bias,
                          float* __restrict__ dt) {
  int idx = blockIdx.x * blockDim.x + threadIdx.x;
  if (idx >= NROWS * H_HEADS) return;
  int h  = idx & (H_HEADS - 1);
  int bl = idx >> 3;
  const float* row = p + (size_t)bl * TOTAL_OUT + OFF_DT;
  float acc = dt_bias[h];
#pragma unroll 8
  for (int k = 0; k < DT_RANK; ++k) acc += row[k] * W_dt[h * DT_RANK + k];
  float sp = (acc > 20.0f) ? acc : log1pf(__expf(acc));
  sp = fminf(fmaxf(sp, DT_MIN), DT_MAX);
  dt[idx] = sp;
}

// ---------------------------------------------------------------------------
// Sequential selective scan. One block per (batch, head); 256 threads, each
// owning one p-channel with the 16-wide state in registers. Per timestep,
// B/C/dA are staged in LDS by lanes 0..15.
// ---------------------------------------------------------------------------
__global__ void scan_kernel(const float* __restrict__ p,
                            const float* __restrict__ u_conv,
                            const float* __restrict__ dt,
                            const float* __restrict__ A_log,
                            const float* __restrict__ Dskip,
                            float* __restrict__ y) {
  const int b = blockIdx.x / H_HEADS;
  const int h = blockIdx.x % H_HEADS;
  const int pch = threadIdx.x;              // 0..255

  __shared__ float Bs[D_STATE], Cs[D_STATE], dAs[D_STATE];

  float an = 0.0f;
  if (pch < D_STATE) an = -__expf(A_log[h * D_STATE + pch]);

  float hstate[D_STATE];
#pragma unroll
  for (int n = 0; n < D_STATE; ++n) hstate[n] = 0.0f;
  const float dsk = Dskip[h * P_DIM + pch];

  for (int t = 0; t < SEQ; ++t) {
    const size_t bl = (size_t)b * SEQ + t;
    const float dt_t = dt[bl * H_HEADS + h];
    if (pch < D_STATE) {
      const float* prow = p + bl * TOTAL_OUT;
      Bs[pch]  = prow[OFF_B + h * D_STATE + pch];
      Cs[pch]  = prow[OFF_C + h * D_STATE + pch];
      dAs[pch] = __expf(dt_t * an);
    }
    __syncthreads();
    const float u_t = u_conv[bl * D_INNER + h * P_DIM + pch];
    const float du  = dt_t * u_t;
    float sum = 0.0f;
#pragma unroll
    for (int n = 0; n < D_STATE; ++n) {
      hstate[n] = hstate[n] * dAs[n] + du * Bs[n];
      sum += hstate[n] * Cs[n];
    }
    y[bl * D_INNER + h * P_DIM + pch] = sum + dsk * u_t;
    __syncthreads();
  }
}

// ---------------------------------------------------------------------------
// Fused gate + residual + RMSNorm, emits bf16 for the output GEMM.
// One block (256 threads) per row; 8 channels per thread.
// ---------------------------------------------------------------------------
__global__ void gate_norm_kernel(const float* __restrict__ p,
                                 const float* __restrict__ y,
                                 const float* __restrict__ res,
                                 const float* __restrict__ norm_w,
                                 __bf16* __restrict__ g_bf) {
  const int bl = blockIdx.x;
  const float* prow = p + (size_t)bl * TOTAL_OUT;       // z at OFF_Z
  const float* yrow = y + (size_t)bl * D_INNER;
  const float* rrow = res + (size_t)bl * D_INNER;

  float g[8];
  float ss = 0.0f;
#pragma unroll
  for (int i = 0; i < 8; ++i) {
    int d = threadIdx.x + i * 256;
    float z = prow[OFF_Z + d];
    float silu = z / (1.0f + __expf(-z));
    float v = yrow[d] * silu + rrow[d];
    g[i] = v;
    ss += v * v;
  }

  __shared__ float red[256];
  red[threadIdx.x] = ss;
  __syncthreads();
  for (int s = 128; s > 0; s >>= 1) {
    if (threadIdx.x < s) red[threadIdx.x] += red[threadIdx.x + s];
    __syncthreads();
  }
  const float scale = rsqrtf(red[0] / (float)D_INNER + EPSV);

  __bf16* grow = g_bf + (size_t)bl * D_INNER;
#pragma unroll
  for (int i = 0; i < 8; ++i) {
    int d = threadIdx.x + i * 256;
    grow[d] = (__bf16)(g[i] * scale * norm_w[d]);
  }
}

// ---------------------------------------------------------------------------
// Launch
// ---------------------------------------------------------------------------
static inline size_t align256(size_t x) { return (x + 255) & ~(size_t)255; }

extern "C" void kernel_launch(void* const* d_in, const int* in_sizes, int n_in,
                              void* d_out, int out_size, void* d_ws, size_t ws_size,
                              hipStream_t stream) {
  const float* x       = (const float*)d_in[0];
  const float* W_in    = (const float*)d_in[1];
  const float* W_dt    = (const float*)d_in[2];
  const float* conv_w  = (const float*)d_in[3];
  const float* conv_b  = (const float*)d_in[4];
  const float* A_log   = (const float*)d_in[5];
  const float* Dskip   = (const float*)d_in[6];
  const float* dt_bias = (const float*)d_in[7];
  const float* norm_w  = (const float*)d_in[8];
  const float* W_out   = (const float*)d_in[9];
  const float* W_res   = (const float*)d_in[10];
  float* out = (float*)d_out;

  // workspace carve-up
  char* ws = (char*)d_ws;
  size_t off = 0;
  __bf16* x_bf    = (__bf16*)(ws + off); off = align256(off + (size_t)NROWS * D_MODEL * 2);
  __bf16* Win_bf  = (__bf16*)(ws + off); off = align256(off + (size_t)TOTAL_OUT * D_MODEL * 2);
  __bf16* Wres_bf = (__bf16*)(ws + off); off = align256(off + (size_t)D_INNER * D_MODEL * 2);
  __bf16* Wout_bf = (__bf16*)(ws + off); off = align256(off + (size_t)D_MODEL * D_INNER * 2);
  float*  p_buf   = (float*)(ws + off);  off = align256(off + (size_t)NROWS * TOTAL_OUT * 4);
  float*  res_buf = (float*)(ws + off);  off = align256(off + (size_t)NROWS * D_INNER * 4);
  float*  u_conv  = (float*)(ws + off);  off = align256(off + (size_t)NROWS * D_INNER * 4);
  float*  dt_buf  = (float*)(ws + off);  off = align256(off + (size_t)NROWS * H_HEADS * 4);
  float*  y_buf   = (float*)(ws + off);  off = align256(off + (size_t)NROWS * D_INNER * 4);
  __bf16* g_bf    = (__bf16*)(ws + off); off = align256(off + (size_t)NROWS * D_INNER * 2);
  (void)ws_size; (void)n_in; (void)in_sizes; (void)out_size;

  // 1) convert operands to bf16
  {
    int n;
    n = NROWS * D_MODEL;
    cvt_f32_bf16_kernel<<<(n + 255) / 256, 256, 0, stream>>>(x, x_bf, n);
    n = TOTAL_OUT * D_MODEL;
    cvt_f32_bf16_kernel<<<(n + 255) / 256, 256, 0, stream>>>(W_in, Win_bf, n);
    n = D_INNER * D_MODEL;
    cvt_f32_bf16_kernel<<<(n + 255) / 256, 256, 0, stream>>>(W_res, Wres_bf, n);
    n = D_MODEL * D_INNER;
    cvt_f32_bf16_kernel<<<(n + 255) / 256, 256, 0, stream>>>(W_out, Wout_bf, n);
  }

  // 2) p = x @ W_in.T   (M=8192, N=4416, K=1024)
  {
    int waves = (NROWS / 64) * (TOTAL_OUT / 32);
    wmma_gemm_bf16_kernel<D_MODEL><<<(waves + 7) / 8, 256, 0, stream>>>(
        x_bf, Win_bf, p_buf, NROWS, TOTAL_OUT);
  }

  // 3) res = x @ W_res.T (M=8192, N=2048, K=1024)
  {
    int waves = (NROWS / 64) * (D_INNER / 32);
    wmma_gemm_bf16_kernel<D_MODEL><<<(waves + 7) / 8, 256, 0, stream>>>(
        x_bf, Wres_bf, res_buf, NROWS, D_INNER);
  }

  // 4) depthwise causal conv on u
  {
    long n = (long)NROWS * D_INNER;
    dwconv_kernel<<<(int)((n + 255) / 256), 256, 0, stream>>>(
        p_buf, conv_w, conv_b, u_conv);
  }

  // 5) dt
  {
    int n = NROWS * H_HEADS;
    dt_kernel<<<(n + 255) / 256, 256, 0, stream>>>(p_buf, W_dt, dt_bias, dt_buf);
  }

  // 6) selective scan
  scan_kernel<<<BATCH * H_HEADS, P_DIM, 0, stream>>>(
      p_buf, u_conv, dt_buf, A_log, Dskip, y_buf);

  // 7) gate + residual + RMSNorm -> bf16
  gate_norm_kernel<<<NROWS, 256, 0, stream>>>(p_buf, y_buf, res_buf, norm_w, g_bf);

  // 8) out = g @ W_out.T (M=8192, N=1024, K=2048), f32 straight to d_out
  {
    int waves = (NROWS / 64) * (D_MODEL / 32);
    wmma_gemm_bf16_kernel<D_INNER><<<(waves + 7) / 8, 256, 0, stream>>>(
        g_bf, Wout_bf, out, NROWS, D_MODEL);
  }
}
